// PairBiasAttention_25855703122343
// MI455X (gfx1250) — matrix operand order repaired and checked by reading, not compiled, not run on hardware
//
#include <hip/hip_runtime.h>
#include <hip/hip_bf16.h>
#include <stdint.h>

// ---------------------------------------------------------------------------
// PairBiasAttention on MI455X (gfx1250): WMMA f16 GEMMs + async-LDS staging.
// B=2 N=2048 K=32 H=8 D=64 NODE_DIM=INNER=512 PAIR_DIM=128
// ---------------------------------------------------------------------------

#define BB 2
#define NN 2048
#define KK 32
#define HH 8
#define DD 64
#define NODE_DIM 512
#define PAIR_DIM 128
#define INNER 512
#define ROWS (BB * NN)          // 4096
#define NQKVG 2048              // [q 512 | k 512 | v 512 | g 512]
#define EPSF 1e-5f

typedef _Float16 v16h __attribute__((ext_vector_type(16)));
typedef _Float16 v8h  __attribute__((ext_vector_type(8)));
typedef float    v8f  __attribute__((ext_vector_type(8)));
typedef int      v4i  __attribute__((ext_vector_type(4)));

// CDNA5 async global->LDS DMA (GLOBAL_LOAD_ASYNC_TO_LDS_B128, ASYNCcnt).
#if defined(__has_builtin)
#  if __has_builtin(__builtin_amdgcn_global_load_async_to_lds_b128) && \
      __has_builtin(__builtin_amdgcn_s_wait_asynccnt)
#    define USE_ASYNC_LDS 1
#  endif
#endif
#ifndef USE_ASYNC_LDS
#  define USE_ASYNC_LDS 0
#endif

#if USE_ASYNC_LDS
// param 0: v4i* in global AS (prints as "int __vector(4) __device__ *")
// param 1: v4i* in LDS AS
typedef __attribute__((address_space(1))) v4i g_v4i;
typedef __attribute__((address_space(3))) v4i l_v4i;
#endif

// ------------------------------- utilities ---------------------------------

__device__ __forceinline__ float wave_sum32(float v) {
  #pragma unroll
  for (int m = 16; m >= 1; m >>= 1) v += __shfl_xor(v, m, 32);
  return v;
}
__device__ __forceinline__ float wave_max32(float v) {
  #pragma unroll
  for (int m = 16; m >= 1; m >>= 1) v = fmaxf(v, __shfl_xor(v, m, 32));
  return v;
}

// --------------------------- 1) node LayerNorm -----------------------------

__global__ void k_ln_node(const float* __restrict__ x,
                          const float* __restrict__ g,
                          const float* __restrict__ b,
                          _Float16* __restrict__ y) {
  const int row = blockIdx.x;
  const int t = threadIdx.x;
  const float* xr = x + (size_t)row * NODE_DIM;
  float x0 = xr[t], x1 = xr[t + 256];

  __shared__ float red[256];
  red[t] = x0 + x1;
  __syncthreads();
  #pragma unroll
  for (int s = 128; s > 0; s >>= 1) { if (t < s) red[t] += red[t + s]; __syncthreads(); }
  const float mu = red[0] * (1.0f / NODE_DIM);
  __syncthreads();
  const float d0 = x0 - mu, d1 = x1 - mu;
  red[t] = d0 * d0 + d1 * d1;
  __syncthreads();
  #pragma unroll
  for (int s = 128; s > 0; s >>= 1) { if (t < s) red[t] += red[t + s]; __syncthreads(); }
  const float rs = rsqrtf(red[0] * (1.0f / NODE_DIM) + EPSF);

  _Float16* yr = y + (size_t)row * NODE_DIM;
  yr[t]       = (_Float16)(d0 * rs * g[t]       + b[t]);
  yr[t + 256] = (_Float16)(d1 * rs * g[t + 256] + b[t + 256]);
}

// ----------------------- 2) pack weights (f16, N-major) --------------------

__global__ void k_pack_qkvg(const float* __restrict__ w_qkv,
                            const float* __restrict__ w_g,
                            const float* __restrict__ b_qkv,
                            const float* __restrict__ b_g,
                            _Float16* __restrict__ Wt,   // [2048][512]
                            float* __restrict__ bias) {  // [2048]
  const int idx = blockIdx.x * 256 + threadIdx.x;        // over 2048*512
  const int nc = idx >> 9;
  const int kc = idx & 511;
  const float v = (nc < 3 * INNER) ? w_qkv[kc * (3 * INNER) + nc]
                                   : w_g[kc * INNER + (nc - 3 * INNER)];
  Wt[idx] = (_Float16)v;
  if (idx < NQKVG)
    bias[idx] = (idx < 3 * INNER) ? b_qkv[idx] : b_g[idx - 3 * INNER];
}

__global__ void k_pack_wout(const float* __restrict__ w_out,
                            _Float16* __restrict__ Wt) { // [512][512]
  const int idx = blockIdx.x * 256 + threadIdx.x;        // over 512*512
  const int nc = idx >> 9;
  const int kc = idx & 511;
  Wt[idx] = (_Float16)w_out[kc * NODE_DIM + nc];
}

// ------------------------- 3) WMMA GEMM (f16 -> f32) -----------------------
// C[M,N] = A[M,Kd] @ Wt[N,Kd]^T + bias[N]
// block tile 64x256, 8 waves = 4(M) x 2(N), wave tile 16x128 (8 wmma tiles).
// Double-buffered LDS; staging via GLOBAL_LOAD_ASYNC_TO_LDS_B128 when the
// toolchain exposes it (ASYNCcnt + barrier), else batched sync loads.

#define A_STRIDE 40
#define B_STRIDE 40
#define A_TILE (64 * A_STRIDE)
#define B_TILE (256 * B_STRIDE)

__device__ __forceinline__ void gemm_stage(const _Float16* __restrict__ A,
                                           const _Float16* __restrict__ Wt,
                                           _Float16* sAbuf, _Float16* sBbuf,
                                           int blockM, int blockN, int Kd,
                                           int kk, int t) {
  const int arow = t >> 2, ac = t & 3;          // A: 256 x 16B chunks
  const _Float16* asrc = A + (size_t)(blockM + arow) * Kd + kk + ac * 8;
  _Float16* adst = sAbuf + arow * A_STRIDE + ac * 8;
#if USE_ASYNC_LDS
  __builtin_amdgcn_global_load_async_to_lds_b128(
      (g_v4i*)asrc, (l_v4i*)adst, 0, 0);
  #pragma unroll
  for (int i = 0; i < 4; ++i) {                 // B: 1024 chunks, 4/thread
    const int chunk = t + i * 256;
    const int brow = chunk >> 2, bc = chunk & 3;
    const _Float16* bsrc = Wt + (size_t)(blockN + brow) * Kd + kk + bc * 8;
    _Float16* bdst = sBbuf + brow * B_STRIDE + bc * 8;
    __builtin_amdgcn_global_load_async_to_lds_b128(
        (g_v4i*)bsrc, (l_v4i*)bdst, 0, 0);
  }
#else
  // Batched fallback: all loads issued into distinct regs before any store.
  uint4 ra = *(const uint4*)asrc;
  uint4 rb[4];
  #pragma unroll
  for (int i = 0; i < 4; ++i) {
    const int chunk = t + i * 256;
    const int brow = chunk >> 2, bc = chunk & 3;
    rb[i] = *(const uint4*)(Wt + (size_t)(blockN + brow) * Kd + kk + bc * 8);
  }
  *(uint4*)adst = ra;
  #pragma unroll
  for (int i = 0; i < 4; ++i) {
    const int chunk = t + i * 256;
    const int brow = chunk >> 2, bc = chunk & 3;
    *(uint4*)(sBbuf + brow * B_STRIDE + bc * 8) = rb[i];
  }
#endif
}

__global__ void __launch_bounds__(256)
k_gemm_wmma(const _Float16* __restrict__ A,
            const _Float16* __restrict__ Wt,
            const float* __restrict__ bias,
            float* __restrict__ C,
            int Mrows, int Ncols, int Kd) {
  __shared__ __align__(16) _Float16 sA[2 * A_TILE];
  __shared__ __align__(16) _Float16 sB[2 * B_TILE];

  const int t = threadIdx.x;
  const int wave = t >> 5;
  const int lane = t & 31;
  const int mL = lane & 15;        // M (A frag) / N (B frag, C) lane index
  const int halfk = lane >> 4;     // lane-group half select
  const int wm = (wave & 3) * 16;  // wave M offset within block tile
  const int wn = (wave >> 2) * 128;// wave N offset within block tile
  const int blockM = blockIdx.x * 64;
  const int blockN = blockIdx.y * 256;

  v8f acc[8];
  const v8f vzero = {0.f, 0.f, 0.f, 0.f, 0.f, 0.f, 0.f, 0.f};
  #pragma unroll
  for (int i = 0; i < 8; ++i) acc[i] = vzero;

  // prologue: stage first K-slice into buffer 0
  gemm_stage(A, Wt, sA, sB, blockM, blockN, Kd, 0, t);

  for (int kk = 0, it = 0; kk < Kd; kk += 32, ++it) {
    const int cur = it & 1;
#if USE_ASYNC_LDS
    __builtin_amdgcn_s_wait_asynccnt(0);   // own DMAs for `cur` done
#endif
    __syncthreads();                       // cur visible; prev reads retired
    if (kk + 32 < Kd)                      // overlap next DMA with compute
      gemm_stage(A, Wt, sA + (cur ^ 1) * A_TILE, sB + (cur ^ 1) * B_TILE,
                 blockM, blockN, Kd, kk + 32, t);

    // ---- A fragment: lane m = lane%16; K = half*8+{0..7}, then +16 ----
    const _Float16* ap = sA + cur * A_TILE + (wm + mL) * A_STRIDE + halfk * 8;
    const v8h alo = *(const v8h*)ap;
    const v8h ahi = *(const v8h*)(ap + 16);
    const v16h afrag = __builtin_shufflevector(alo, ahi,
        0, 1, 2, 3, 4, 5, 6, 7, 8, 9, 10, 11, 12, 13, 14, 15);

    // ---- 8 N tiles: B frag lane n = lane%16; K = half*8+{0..7}, then +16 ----
    const _Float16* bbase = sB + cur * B_TILE;
    #pragma unroll
    for (int tt = 0; tt < 8; ++tt) {
      const _Float16* bp = bbase + (wn + tt * 16 + mL) * B_STRIDE + halfk * 8;
      const v8h b0 = *(const v8h*)bp;
      const v8h b1 = *(const v8h*)(bp + 16);
      const v16h bfrag = __builtin_shufflevector(b0, b1,
          0, 1, 2, 3, 4, 5, 6, 7, 8, 9, 10, 11, 12, 13, 14, 15);
      acc[tt] = __builtin_amdgcn_wmma_f32_16x16x32_f16(
          false, afrag, false, bfrag, (short)0, acc[tt], false, false);
    }
  }

  // ---- epilogue: C VGPR r -> M = half*8 + r, N = lane%16 ----
  #pragma unroll
  for (int tt = 0; tt < 8; ++tt) {
    const int col = blockN + wn + tt * 16 + mL;
    const float bv = bias[col];
    #pragma unroll
    for (int r = 0; r < 8; ++r) {
      const int rowg = blockM + wm + halfk * 8 + r;
      C[(size_t)rowg * Ncols + col] = acc[tt][r] + bv;
    }
  }
}

// ------------------- 4) LayerNorm q / k slices (in place) ------------------

__global__ void k_ln_qk(float* __restrict__ qkvg,
                        const float* __restrict__ gq, const float* __restrict__ bq,
                        const float* __restrict__ gk, const float* __restrict__ bk) {
  const int row = blockIdx.x;
  const int seg = blockIdx.y;                       // 0 = q, 1 = k
  const int t = threadIdx.x;
  float* xr = qkvg + (size_t)row * NQKVG + seg * INNER;
  const float* g = seg ? gk : gq;
  const float* b = seg ? bk : bq;

  float x0 = xr[t], x1 = xr[t + 256];
  __shared__ float red[256];
  red[t] = x0 + x1;
  __syncthreads();
  #pragma unroll
  for (int s = 128; s > 0; s >>= 1) { if (t < s) red[t] += red[t + s]; __syncthreads(); }
  const float mu = red[0] * (1.0f / INNER);
  __syncthreads();
  const float d0 = x0 - mu, d1 = x1 - mu;
  red[t] = d0 * d0 + d1 * d1;
  __syncthreads();
  #pragma unroll
  for (int s = 128; s > 0; s >>= 1) { if (t < s) red[t] += red[t + s]; __syncthreads(); }
  const float rs = rsqrtf(red[0] * (1.0f / INNER) + EPSF);
  xr[t]       = d0 * rs * g[t]       + b[t];
  xr[t + 256] = d1 * rs * g[t + 256] + b[t + 256];
}

// --------------- 5) pair LayerNorm + bias einsum (wave per row) ------------

__global__ void k_pair_bias(const float* __restrict__ pair,
                            const float* __restrict__ w_bias,   // [128][8]
                            const float* __restrict__ g,
                            const float* __restrict__ b,
                            float* __restrict__ biasHNK) {
  const int wave = threadIdx.x >> 5;
  const int lane = threadIdx.x & 31;
  const int row = blockIdx.x * 8 + wave;            // over B*N*K = 131072
  const int b_  = row >> 16;                        // / (N*K)
  const int rem = row & 65535;
  const int n   = rem >> 5;
  const int j   = rem & 31;

  const float4 x4 = ((const float4*)(pair + (size_t)row * PAIR_DIM))[lane];
  float xv[4] = {x4.x, x4.y, x4.z, x4.w};

  float s = xv[0] + xv[1] + xv[2] + xv[3];
  const float mu = wave_sum32(s) * (1.0f / PAIR_DIM);
  float sq = 0.f;
  #pragma unroll
  for (int i = 0; i < 4; ++i) { xv[i] -= mu; sq += xv[i] * xv[i]; }
  const float rs = rsqrtf(wave_sum32(sq) * (1.0f / PAIR_DIM) + EPSF);

  float xn[4];
  #pragma unroll
  for (int i = 0; i < 4; ++i) {
    const int p = lane * 4 + i;
    xn[i] = xv[i] * rs * g[p] + b[p];
  }

  float* dst = biasHNK + (((size_t)b_ * HH) * NN + n) * KK + j;
  #pragma unroll
  for (int h = 0; h < HH; ++h) {
    float part = 0.f;
    #pragma unroll
    for (int i = 0; i < 4; ++i) part += xn[i] * w_bias[(lane * 4 + i) * HH + h];
    const float tot = wave_sum32(part);
    if (lane == 0) dst[(size_t)h * NN * KK] = tot;
  }
}

// -------------- 6) neighbor attention + gate (wave per head) ---------------

__global__ void k_attn(const float* __restrict__ qkvg,
                       const float* __restrict__ biasHNK,
                       const int* __restrict__ nbr,
                       const unsigned char* __restrict__ mask,
                       _Float16* __restrict__ gated) {
  const int bid = blockIdx.x;                       // b*N + n
  const int b_  = bid >> 11;
  const int n   = bid & 2047;
  const int h   = threadIdx.x >> 5;
  const int lane = threadIdx.x & 31;

  const int idx = nbr[(size_t)bid * KK + lane];
  const bool valid = (mask[(size_t)b_ * NN + idx] != 0) && (mask[bid] != 0);

  // q.k dot with float4 vector loads (rows are 16B aligned)
  const float4* q4 = (const float4*)(qkvg + (size_t)bid * NQKVG + h * DD);
  const float4* k4 = (const float4*)(qkvg + (size_t)(b_ * NN + idx) * NQKVG +
                                     INNER + h * DD);
  float s = 0.f;
  #pragma unroll
  for (int d = 0; d < DD / 4; ++d) {
    const float4 qa = q4[d], kb = k4[d];
    s = fmaf(qa.x, kb.x, fmaf(qa.y, kb.y, fmaf(qa.z, kb.z, fmaf(qa.w, kb.w, s))));
  }
  s *= 0.125f;                                      // D^-0.5

  const float bia = biasHNK[(((size_t)b_ * HH + h) * NN + n) * KK + lane];
  const float sm = (valid ? s : -3.402823466e+38f) + bia;

  const float mx = wave_max32(sm);
  const float e  = __expf(sm - mx);
  const float a  = e / wave_sum32(e);

  float o0 = 0.f, o1 = 0.f;
  const int d0 = lane * 2;
  #pragma unroll 4
  for (int j2 = 0; j2 < KK; ++j2) {
    const float aj = __shfl(a, j2, 32);
    const int   ij = __shfl(idx, j2, 32);
    const float2 vv = *(const float2*)(qkvg + (size_t)(b_ * NN + ij) * NQKVG +
                                       2 * INNER + h * DD + d0);
    o0 = fmaf(aj, vv.x, o0);
    o1 = fmaf(aj, vv.y, o1);
  }

  const float2 gg = *(const float2*)(qkvg + (size_t)bid * NQKVG + 3 * INNER +
                                     h * DD + d0);
  _Float16* orow = gated + (size_t)bid * INNER + h * DD;
  orow[d0]     = (_Float16)(o0 / (1.f + __expf(-gg.x)));
  orow[d0 + 1] = (_Float16)(o1 / (1.f + __expf(-gg.y)));
}

// ------------------------------- launcher ----------------------------------

extern "C" void kernel_launch(void* const* d_in, const int* in_sizes, int n_in,
                              void* d_out, int out_size, void* d_ws, size_t ws_size,
                              hipStream_t stream) {
  const float* node   = (const float*)d_in[0];
  const float* pair   = (const float*)d_in[1];
  const unsigned char* mask = (const unsigned char*)d_in[2];
  const int*   nbr    = (const int*)d_in[3];
  const float* w_qkv  = (const float*)d_in[4];
  const float* b_qkv  = (const float*)d_in[5];
  const float* w_g    = (const float*)d_in[6];
  const float* b_g    = (const float*)d_in[7];
  const float* w_out  = (const float*)d_in[8];
  const float* b_out  = (const float*)d_in[9];
  const float* w_bias = (const float*)d_in[10];
  const float* ln_ng  = (const float*)d_in[11];
  const float* ln_nb  = (const float*)d_in[12];
  const float* ln_qg  = (const float*)d_in[13];
  const float* ln_qb  = (const float*)d_in[14];
  const float* ln_kg  = (const float*)d_in[15];
  const float* ln_kb  = (const float*)d_in[16];
  const float* ln_pg  = (const float*)d_in[17];
  const float* ln_pb  = (const float*)d_in[18];

  char* p = (char*)d_ws;
  _Float16* A_node   = (_Float16*)p; p += (size_t)ROWS * NODE_DIM * 2;       // 4 MB
  _Float16* Wt_qkvg  = (_Float16*)p; p += (size_t)NQKVG * NODE_DIM * 2;      // 2 MB
  _Float16* Wt_out   = (_Float16*)p; p += (size_t)NODE_DIM * INNER * 2;      // 0.5 MB
  float*    bias2048 = (float*)p;    p += (size_t)NQKVG * 4;                 // 8 KB
  float*    QKVG     = (float*)p;    p += (size_t)ROWS * NQKVG * 4;         // 32 MB
  float*    biasHNK  = (float*)p;    p += (size_t)BB * HH * NN * KK * 4;    // 4 MB
  _Float16* gated    = (_Float16*)p; p += (size_t)ROWS * INNER * 2;          // 4 MB

  k_ln_node<<<ROWS, 256, 0, stream>>>(node, ln_ng, ln_nb, A_node);
  k_pack_qkvg<<<(NQKVG * NODE_DIM) / 256, 256, 0, stream>>>(
      w_qkv, w_g, b_qkv, b_g, Wt_qkvg, bias2048);
  k_pack_wout<<<(NODE_DIM * INNER) / 256, 256, 0, stream>>>(w_out, Wt_out);
  k_gemm_wmma<<<dim3(ROWS / 64, NQKVG / 256), 256, 0, stream>>>(
      A_node, Wt_qkvg, bias2048, QKVG, ROWS, NQKVG, NODE_DIM);
  k_ln_qk<<<dim3(ROWS, 2), 256, 0, stream>>>(QKVG, ln_qg, ln_qb, ln_kg, ln_kb);
  k_pair_bias<<<(BB * NN * KK) / 8, 256, 0, stream>>>(
      pair, w_bias, ln_pg, ln_pb, biasHNK);
  k_attn<<<ROWS, 256, 0, stream>>>(QKVG, biasHNK, nbr, mask, gated);
  k_gemm_wmma<<<dim3(ROWS / 64, INNER / 256), 256, 0, stream>>>(
      gated, Wt_out, b_out, (float*)d_out, ROWS, NODE_DIM, INNER);
}
